// GNN_FBDD_2044404432995
// MI455X (gfx1250) — matrix operand order repaired and checked by reading, not compiled, not run on hardware
//
#include <hip/hip_runtime.h>

// ---------------------------------------------------------------------------
// Types for CDNA5 WMMA (wave32). v_wmma_f32_16x16x32_bf16:
//   (bool neg_a, v16bf A, bool neg_b, v16bf B, short c_mod, v8f C, bool rA, bool rB)
// ---------------------------------------------------------------------------
typedef __bf16 bf16_t;
typedef bf16_t v16bf __attribute__((ext_vector_type(16)));
typedef float  v8f   __attribute__((ext_vector_type(8)));
typedef int    v4i   __attribute__((ext_vector_type(4)));

#define WAVE 32

// Async global->LDS staging (GLOBAL_LOAD_ASYNC_TO_LDS_B128, ASYNCcnt) if the
// toolchain exposes it; otherwise fall back to synchronous b128 staging.
#if defined(__AMDGCN__) && __has_builtin(__builtin_amdgcn_global_load_async_to_lds_b128) && __has_builtin(__builtin_amdgcn_s_wait_asynccnt)
#define USE_ASYNC_LDS 1
typedef __attribute__((address_space(1))) v4i* gv4i_p;
typedef __attribute__((address_space(3))) v4i* lv4i_p;
#else
#define USE_ASYNC_LDS 0
#endif

__device__ __forceinline__ float gelu_exact(float x) {
    // 0.5*x*(1+erf(x/sqrt(2)))  (matches jax.nn.gelu approximate=False)
    return 0.5f * x * (1.0f + erff(x * 0.70710678118654752440f));
}
__device__ __forceinline__ float sigmoidf_(float x) { return 1.0f / (1.0f + expf(-x)); }

__device__ __forceinline__ float wave_sum(float v) {
#pragma unroll
    for (int off = 16; off > 0; off >>= 1) v += __shfl_xor(v, off, WAVE);
    return v;
}

// Order-preserving float<->uint encoding for atomicMax-based segment_max.
__device__ __forceinline__ unsigned f2o(float f) {
    unsigned u = __float_as_uint(f);
    return (u & 0x80000000u) ? ~u : (u | 0x80000000u);
}
__device__ __forceinline__ float o2f(unsigned u) {
    return (u & 0x80000000u) ? __uint_as_float(u & 0x7FFFFFFFu) : __uint_as_float(~u);
}
#define NEG_INF_ORD 0x007FFFFFu   // f2o(-inf)

// ---------------------------------------------------------------------------
// Weight pre-pack into the per-lane B-fragment layout of
// v_wmma_f32_16x16x32_bf16: fragment element i of lane L =
//   W[k0 + ((L<16)?0:8) + ((i<8)?i:8+i)][ncol],  ncol = tn*16 + (L&15).
// transb=1 reads W as [Nd x K] row-major (for A@W^T).
// Linear layout: [(tn*KS + ks)*32 + lane] * 16 bf16  (32B/lane, coalesced).
// ---------------------------------------------------------------------------
__global__ __launch_bounds__(256) void pack_b_kernel(
    const float* __restrict__ W, bf16_t* __restrict__ out,
    int Nd, int K, int transb, long total /* (Nd/16)*KS*32 */)
{
    long idx = (long)blockIdx.x * blockDim.x + threadIdx.x;
    if (idx >= total) return;
    int  lane = (int)(idx & 31);
    long t    = idx >> 5;
    int  KS   = (K + 31) / 32;
    int  ks   = (int)(t % KS);
    int  tn   = (int)(t / KS);
    int  ncol = tn * 16 + (lane & 15);
    int  kbB  = (lane < 16) ? 0 : 8;
    v16bf v;
#pragma unroll
    for (int i = 0; i < 16; ++i) {
        int kb = ks * 32 + kbB + ((i < 8) ? i : 8 + i);
        float f = 0.f;
        if (kb < K)
            f = transb ? W[(size_t)ncol * K + kb] : W[(size_t)kb * Nd + ncol];
        v[i] = (bf16_t)f;
    }
    *(v16bf*)(out + (size_t)idx * 16) = v;
}

// Build an A fragment (v16bf) for one 32-wide k-step from a row pointer.
// Lane layout: elements 0..7 = rows [k0+kbA, +8), 8..15 = rows [k0+16+kbA, +8).
template <int K>
__device__ __forceinline__ v16bf load_a_frag(const float* __restrict__ arow,
                                             int k0, int kbA)
{
    float4 a0 = *(const float4*)(arow + k0 + kbA);
    float4 a1 = *(const float4*)(arow + k0 + kbA + 4);
    float4 a2, a3;
    if (k0 + 16 < K) {                 // compile-time resolved (K=16 case)
        a2 = *(const float4*)(arow + k0 + 16 + kbA);
        a3 = *(const float4*)(arow + k0 + 16 + kbA + 4);
    } else {
        a2 = make_float4(0.f, 0.f, 0.f, 0.f);
        a3 = make_float4(0.f, 0.f, 0.f, 0.f);
    }
    v16bf af;
    af[0]  = (bf16_t)a0.x; af[1]  = (bf16_t)a0.y; af[2]  = (bf16_t)a0.z; af[3]  = (bf16_t)a0.w;
    af[4]  = (bf16_t)a1.x; af[5]  = (bf16_t)a1.y; af[6]  = (bf16_t)a1.z; af[7]  = (bf16_t)a1.w;
    af[8]  = (bf16_t)a2.x; af[9]  = (bf16_t)a2.y; af[10] = (bf16_t)a2.z; af[11] = (bf16_t)a2.w;
    af[12] = (bf16_t)a3.x; af[13] = (bf16_t)a3.y; af[14] = (bf16_t)a3.z; af[15] = (bf16_t)a3.w;
    return af;
}

// ---------------------------------------------------------------------------
// Specialized WMMA GEMM for Nd=128:  C[M x 128] = act((A [+A2]) @ W + bias).
// Block = 256 threads = 8 waves = one 16-row block x all 8 N-tiles.
// The 16 x K A panel is staged once in LDS and shared by all 8 waves
// (8x fewer global A reads). Staging uses the gfx1250 async global->LDS
// path (ASYNCcnt) when no A2 fusion is needed. LDS rows padded to K+4
// floats so the 16 lanes reading 16 rows hit distinct bank groups.
// ---------------------------------------------------------------------------
template <int ACT, int K, bool USE_A2>
__global__ __launch_bounds__(256) void wmma_gemm128_kernel(
    const float* __restrict__ A, const float* __restrict__ A2,
    const bf16_t* __restrict__ Bpk, const float* __restrict__ bias,
    float* __restrict__ C)
{
    constexpr int KS   = (K + 31) / 32;
    constexpr int SROW = K + 4;              // padded LDS row stride (floats)
    __shared__ float sA[16 * SROW];

    const int tm  = blockIdx.x;
    const int tid = threadIdx.x;

    // ---- stage the 16 x K panel into LDS ----
    constexpr int NV4 = 16 * K / 4;          // float4 chunks
    const float* __restrict__ Ablk = A + (size_t)tm * 16 * K;
#if USE_ASYNC_LDS
    if (!USE_A2) {
        for (int i = tid; i < NV4; i += 256) {
            int row = i / (K / 4);
            int cg  = i - row * (K / 4);
            __builtin_amdgcn_global_load_async_to_lds_b128(
                (gv4i_p)(void*)(Ablk + (size_t)row * K + cg * 4),
                (lv4i_p)(sA + row * SROW + cg * 4),
                0, 0);
        }
        __builtin_amdgcn_s_wait_asynccnt(0);
    } else
#endif
    {
        for (int i = tid; i < NV4; i += 256) {
            int row = i / (K / 4);
            int cg  = i - row * (K / 4);
            float4 v = *(const float4*)(Ablk + (size_t)row * K + cg * 4);
            if (USE_A2) {
                float4 w = *(const float4*)(A2 + (size_t)tm * 16 * K + (size_t)row * K + cg * 4);
                v.x += w.x; v.y += w.y; v.z += w.z; v.w += w.w;
            }
            *(float4*)(sA + row * SROW + cg * 4) = v;
        }
    }
    __syncthreads();

    // ---- per-wave 16x16 tile ----
    const int lane = tid & 31;
    const int tn   = tid >> 5;               // 0..7 (= N-tile)
    const int ncol = tn * 16 + (lane & 15);
    const int kbA  = (lane < 16) ? 0 : 8;
    const float*  __restrict__ arow = sA + (lane & 15) * SROW;
    const bf16_t* __restrict__ bp   = Bpk + ((size_t)tn * KS * 32 + lane) * 16;

    v8f acc = {0.f, 0.f, 0.f, 0.f, 0.f, 0.f, 0.f, 0.f};
#pragma unroll
    for (int ks = 0; ks < KS; ++ks) {
        v16bf af  = load_a_frag<K>(arow, ks * 32, kbA);
        v16bf bfv = *(const v16bf*)(bp + (size_t)ks * 32 * 16);
        acc = __builtin_amdgcn_wmma_f32_16x16x32_bf16(false, af, false, bfv,
                                                      (short)0, acc, false, false);
    }

    const float bcol  = bias[ncol];
    const int   rbase = tm * 16 + ((lane < 16) ? 0 : 8);
#pragma unroll
    for (int i = 0; i < 8; ++i) {
        float v = acc[i] + bcol;
        if (ACT == 1) v = gelu_exact(v);
        C[(size_t)(rbase + i) * 128 + ncol] = v;
    }
}

// ---------------------------------------------------------------------------
// Generic WMMA GEMM (register-only A), used for the small Nd=512 LSTM GEMMs.
//   ACC=1: C += product + bias.
// ---------------------------------------------------------------------------
template <int ACT, bool ACC, int K>
__global__ __launch_bounds__(256) void wmma_gemm_kernel(
    const float* __restrict__ A, const bf16_t* __restrict__ Bpk,
    const float* __restrict__ bias, float* __restrict__ C, int M, int Nd)
{
    constexpr int KS = (K + 31) / 32;
    const int lane = threadIdx.x & (WAVE - 1);
    const int wave = blockIdx.x * (blockDim.x >> 5) + (threadIdx.x >> 5);
    const int ntile = Nd >> 4;
    const int tm = wave / ntile;
    const int tn = wave - tm * ntile;
    if (tm * 16 >= M) return;

    const int mrow = tm * 16 + (lane & 15);
    const int ncol = tn * 16 + (lane & 15);
    const int kbA  = (lane < 16) ? 0 : 8;

    const float*  __restrict__ arow = A + (size_t)mrow * K;
    const bf16_t* __restrict__ bp   = Bpk + ((size_t)tn * KS * 32 + lane) * 16;

    v8f acc = {0.f, 0.f, 0.f, 0.f, 0.f, 0.f, 0.f, 0.f};
#pragma unroll
    for (int ks = 0; ks < KS; ++ks) {
        v16bf af  = load_a_frag<K>(arow, ks * 32, kbA);
        v16bf bfv = *(const v16bf*)(bp + (size_t)ks * 32 * 16);
        acc = __builtin_amdgcn_wmma_f32_16x16x32_bf16(false, af, false, bfv,
                                                      (short)0, acc, false, false);
    }

    const float bcol  = bias ? bias[ncol] : 0.f;
    const int   rbase = tm * 16 + ((lane < 16) ? 0 : 8);
#pragma unroll
    for (int i = 0; i < 8; ++i) {
        int row = rbase + i;
        float v = acc[i] + bcol;
        if (ACC) v += C[(size_t)row * Nd + ncol];
        if (ACT == 1) v = gelu_exact(v);
        C[(size_t)row * Nd + ncol] = v;
    }
}

// ---------------------------------------------------------------------------
// GINE message + scatter: agg[dst] += relu(h[src] + e)   (per edge element)
// ---------------------------------------------------------------------------
__global__ __launch_bounds__(256) void gine_msg_kernel(
    const float* __restrict__ h, const float* __restrict__ e,
    const int* __restrict__ src, const int* __restrict__ dst,
    float* __restrict__ agg, long total /* E*128 */)
{
    long idx = (long)blockIdx.x * blockDim.x + threadIdx.x;
    if (idx >= total) return;
    int ed = (int)(idx >> 7);
    int c  = (int)(idx & 127);
    int s = src[ed], d = dst[ed];
    float m = h[(size_t)s * 128 + c] + e[idx];
    m = fmaxf(m, 0.f);
    atomicAdd(&agg[(size_t)d * 128 + c], m);
}

// ---------------------------------------------------------------------------
// h = gelu(LayerNorm(zz; g,b)) + h   (wave per row, H=128, 4 vals per lane)
// ---------------------------------------------------------------------------
__global__ __launch_bounds__(256) void ln_gelu_res_kernel(
    const float* __restrict__ zz, const float* __restrict__ g,
    const float* __restrict__ b, float* __restrict__ h, int M)
{
    int wave = blockIdx.x * (blockDim.x >> 5) + (threadIdx.x >> 5);
    int lane = threadIdx.x & 31;
    if (wave >= M) return;
    size_t base = (size_t)wave * 128;
    float x[4];
    float s = 0.f;
#pragma unroll
    for (int i = 0; i < 4; ++i) { x[i] = zz[base + lane + 32 * i]; s += x[i]; }
    s = wave_sum(s);
    float mu = s * (1.0f / 128.0f);
    float vs = 0.f;
#pragma unroll
    for (int i = 0; i < 4; ++i) { float d = x[i] - mu; vs += d * d; }
    vs = wave_sum(vs);
    float rinv = rsqrtf(vs * (1.0f / 128.0f) + 1e-5f);
#pragma unroll
    for (int i = 0; i < 4; ++i) {
        int c = lane + 32 * i;
        float ln = (x[i] - mu) * rinv * g[c] + b[c];
        h[base + c] = gelu_exact(ln) + h[base + c];
    }
}

// ---------------------------------------------------------------------------
// Attention: alpha[e][head] = dot32(q[dst], k[src]+ee)/sqrt(C); segment max.
// ---------------------------------------------------------------------------
__global__ __launch_bounds__(256) void attn_alpha_kernel(
    const float* __restrict__ q, const float* __restrict__ k,
    const float* __restrict__ ee, const int* __restrict__ src,
    const int* __restrict__ dst, float* __restrict__ alpha,
    unsigned* __restrict__ amax, int E)
{
    int wave = blockIdx.x * (blockDim.x >> 5) + (threadIdx.x >> 5);
    int lane = threadIdx.x & 31;
    if (wave >= E) return;
    int s = src[wave], d = dst[wave];
    const float scale = 0.17677669529663688f; // 1/sqrt(32)
#pragma unroll
    for (int hd = 0; hd < 4; ++hd) {
        int c = hd * 32 + lane;
        float v = q[(size_t)d * 128 + c] * (k[(size_t)s * 128 + c] + ee[(size_t)wave * 128 + c]);
        v = wave_sum(v);
        if (lane == 0) {
            float a = v * scale;
            alpha[(size_t)wave * 4 + hd] = a;
            atomicMax(&amax[(size_t)d * 4 + hd], f2o(a));
        }
    }
}

// ealpha = exp(alpha - amax[dst]); denom[dst] += ealpha   (in-place on alpha)
__global__ __launch_bounds__(256) void attn_exp_kernel(
    float* __restrict__ alpha, const unsigned* __restrict__ amax,
    float* __restrict__ denom, const int* __restrict__ dst, long total /*E*4*/)
{
    long idx = (long)blockIdx.x * blockDim.x + threadIdx.x;
    if (idx >= total) return;
    int ed = (int)(idx >> 2);
    int hd = (int)(idx & 3);
    int d = dst[ed];
    float ea = expf(alpha[idx] - o2f(amax[(size_t)d * 4 + hd]));
    alpha[idx] = ea;
    atomicAdd(&denom[(size_t)d * 4 + hd], ea);
}

// out[dst] += attn * (v[src] + ee)
__global__ __launch_bounds__(256) void attn_agg_kernel(
    const float* __restrict__ ealpha, const float* __restrict__ denom,
    const float* __restrict__ v, const float* __restrict__ ee,
    const int* __restrict__ src, const int* __restrict__ dst,
    float* __restrict__ outb, long total /*E*128*/)
{
    long idx = (long)blockIdx.x * blockDim.x + threadIdx.x;
    if (idx >= total) return;
    int ed = (int)(idx >> 7);
    int c  = (int)(idx & 127);
    int hd = c >> 5;
    int s = src[ed], d = dst[ed];
    float attn = ealpha[(size_t)ed * 4 + hd] / (denom[(size_t)d * 4 + hd] + 1e-16f);
    atomicAdd(&outb[(size_t)d * 128 + c], attn * (v[(size_t)s * 128 + c] + ee[idx]));
}

// Beta gate + LayerNorm: h = LN(h + g*r + (1-g)*out), g = sigmoid([out,r,out-r]@Wbeta)
__global__ __launch_bounds__(256) void beta_ln_kernel(
    const float* __restrict__ ob, const float* __restrict__ rr,
    const float* __restrict__ wbeta, const float* __restrict__ tg,
    const float* __restrict__ tb, float* __restrict__ h, int M)
{
    int wave = blockIdx.x * (blockDim.x >> 5) + (threadIdx.x >> 5);
    int lane = threadIdx.x & 31;
    if (wave >= M) return;
    size_t base = (size_t)wave * 128;
    float o[4], r4[4], hh[4];
    float dotv = 0.f;
#pragma unroll
    for (int i = 0; i < 4; ++i) {
        int c = lane + 32 * i;
        o[i]  = ob[base + c];
        r4[i] = rr[base + c];
        hh[i] = h[base + c];
        dotv += o[i] * wbeta[c] + r4[i] * wbeta[128 + c] + (o[i] - r4[i]) * wbeta[256 + c];
    }
    dotv = wave_sum(dotv);
    float g = sigmoidf_(dotv);
    float x[4];
    float s = 0.f;
#pragma unroll
    for (int i = 0; i < 4; ++i) { x[i] = hh[i] + g * r4[i] + (1.f - g) * o[i]; s += x[i]; }
    s = wave_sum(s);
    float mu = s * (1.0f / 128.0f);
    float vs = 0.f;
#pragma unroll
    for (int i = 0; i < 4; ++i) { float d = x[i] - mu; vs += d * d; }
    vs = wave_sum(vs);
    float rinv = rsqrtf(vs * (1.0f / 128.0f) + 1e-5f);
#pragma unroll
    for (int i = 0; i < 4; ++i) {
        int c = lane + 32 * i;
        h[base + c] = (x[i] - mu) * rinv * tg[c] + tb[c];
    }
}

// ---------------------------------------------------------------------------
// Set2Set
// ---------------------------------------------------------------------------
__global__ __launch_bounds__(256) void lstm_update_kernel(
    const float* __restrict__ gates, float* __restrict__ cs,
    float* __restrict__ hs, int total /* B*128 */)
{
    int idx = blockIdx.x * blockDim.x + threadIdx.x;
    if (idx >= total) return;
    int b = idx >> 7, c = idx & 127;
    const float* gr = gates + (size_t)b * 512;
    float ig = sigmoidf_(gr[c]);
    float fg = sigmoidf_(gr[128 + c]);
    float gg = tanhf(gr[256 + c]);
    float og = sigmoidf_(gr[384 + c]);
    float cn = fg * cs[idx] + ig * gg;
    cs[idx] = cn;
    hs[idx] = og * tanhf(cn);
}

__global__ __launch_bounds__(256) void s2s_score_kernel(
    const float* __restrict__ h, const float* __restrict__ hs,
    const int* __restrict__ batch, float* __restrict__ score,
    unsigned* __restrict__ smax, int Nn)
{
    int wave = blockIdx.x * (blockDim.x >> 5) + (threadIdx.x >> 5);
    int lane = threadIdx.x & 31;
    if (wave >= Nn) return;
    int bt = batch[wave];
    float s = 0.f;
#pragma unroll
    for (int i = 0; i < 4; ++i) {
        int c = lane + 32 * i;
        s += h[(size_t)wave * 128 + c] * hs[(size_t)bt * 128 + c];
    }
    s = wave_sum(s);
    if (lane == 0) {
        score[wave] = s;
        atomicMax(&smax[bt], f2o(s));
    }
}

__global__ __launch_bounds__(256) void s2s_exp_kernel(
    float* __restrict__ score, const unsigned* __restrict__ smax,
    float* __restrict__ ssum, const int* __restrict__ batch, int Nn)
{
    int n = blockIdx.x * blockDim.x + threadIdx.x;
    if (n >= Nn) return;
    int bt = batch[n];
    float e = expf(score[n] - o2f(smax[bt]));
    score[n] = e;
    atomicAdd(&ssum[bt], e);
}

__global__ __launch_bounds__(256) void s2s_pool_kernel(
    const float* __restrict__ score, const float* __restrict__ ssum,
    const float* __restrict__ h, const int* __restrict__ batch,
    float* __restrict__ rpool, long total /* N*128 */)
{
    long idx = (long)blockIdx.x * blockDim.x + threadIdx.x;
    if (idx >= total) return;
    int n = (int)(idx >> 7), c = (int)(idx & 127);
    int bt = batch[n];
    float a = score[n] / (ssum[bt] + 1e-16f);
    atomicAdd(&rpool[(size_t)bt * 128 + c], a * h[idx]);
}

__global__ __launch_bounds__(256) void pack_qstar_kernel(
    const float* __restrict__ hs, const float* __restrict__ rpool,
    float* __restrict__ qstar, int total /* B*256 */)
{
    int idx = blockIdx.x * blockDim.x + threadIdx.x;
    if (idx >= total) return;
    int b = idx >> 8, c = idx & 255;
    qstar[idx] = (c < 128) ? hs[(size_t)b * 128 + c] : rpool[(size_t)b * 128 + (c - 128)];
}

__global__ __launch_bounds__(256) void head_out_kernel(
    const float* __restrict__ g1, const float* __restrict__ W2,
    const float* __restrict__ b2, float* __restrict__ out, int Bn)
{
    int b = blockIdx.x * blockDim.x + threadIdx.x;
    if (b >= Bn) return;
    float s = b2[0];
#pragma unroll 8
    for (int c = 0; c < 128; ++c) s += g1[(size_t)b * 128 + c] * W2[c];
    out[b] = s;
}

__global__ __launch_bounds__(256) void fill_u32_kernel(unsigned* p, unsigned v, long n) {
    long i = (long)blockIdx.x * blockDim.x + threadIdx.x;
    if (i < n) p[i] = v;
}

// ---------------------------------------------------------------------------
// Host side
// ---------------------------------------------------------------------------
static inline int cdiv(long a, long b) { return (int)((a + b - 1) / b); }

extern "C" void kernel_launch(void* const* d_in, const int* in_sizes, int n_in,
                              void* d_out, int out_size, void* d_ws, size_t ws_size,
                              hipStream_t stream)
{
    const int FEAT = 64, EDGE = 16, H = 128, HEADS = 4, B = 512, L = 3, STEPS = 3;
    const int N = in_sizes[0] / FEAT;
    const int E = in_sizes[1] / 2;

    const float* x      = (const float*)d_in[0];
    const int*   eidx   = (const int*)d_in[1];
    const float* eattr  = (const float*)d_in[2];
    const int*   batch  = (const int*)d_in[3];
    const float* W_node = (const float*)d_in[4];
    const float* b_node = (const float*)d_in[5];
    const float* W_edge = (const float*)d_in[6];
    const float* b_edge = (const float*)d_in[7];
    const float* mlp_W1 = (const float*)d_in[8];
    const float* mlp_b1 = (const float*)d_in[9];
    const float* mlp_W2 = (const float*)d_in[10];
    const float* mlp_b2 = (const float*)d_in[11];
    const float* ln_g   = (const float*)d_in[12];
    const float* ln_b   = (const float*)d_in[13];
    const float* Wq     = (const float*)d_in[14];
    const float* bq     = (const float*)d_in[15];
    const float* Wk     = (const float*)d_in[16];
    const float* bk     = (const float*)d_in[17];
    const float* Wv     = (const float*)d_in[18];
    const float* bv     = (const float*)d_in[19];
    const float* We     = (const float*)d_in[20];
    const float* be     = (const float*)d_in[21];
    const float* Wskip  = (const float*)d_in[22];
    const float* bskip  = (const float*)d_in[23];
    const float* Wbeta  = (const float*)d_in[24];
    const float* tln_g  = (const float*)d_in[25];
    const float* tln_b  = (const float*)d_in[26];
    const float* lstm_Wih = (const float*)d_in[27];
    const float* lstm_Whh = (const float*)d_in[28];
    const float* lstm_bih = (const float*)d_in[29];
    const float* lstm_bhh = (const float*)d_in[30];
    const float* head_W1  = (const float*)d_in[31];
    const float* head_b1  = (const float*)d_in[32];
    const float* head_W2  = (const float*)d_in[33];
    const float* head_b2  = (const float*)d_in[34];

    const int* src = eidx;
    const int* dst = eidx + E;

    // -------- workspace layout (floats) --------
    const size_t NH = (size_t)N * H;
    const size_t EH = (size_t)E * H;
    float* ws = (float*)d_ws;
    size_t off = 0;
    float* h     = ws + off; off += NH;
    float* agg   = ws + off; off += NH;   // GINE aggregate; reused as attention output
    float* t1    = ws + off; off += NH;
    float* zz    = ws + off; off += NH;
    float* qb    = ws + off; off += NH;
    float* kb    = ws + off; off += NH;
    float* vb    = ws + off; off += NH;
    float* rb    = ws + off; off += NH;
    float* ebuf  = ws + off; off += EH;
    float* eeb   = ws + off; off += EH;
    float* alpha = ws + off; off += (size_t)E * HEADS;
    float* denom = ws + off; off += (size_t)N * HEADS;
    unsigned* amax = (unsigned*)(ws + off); off += (size_t)N * HEADS;
    float* gates = ws + off; off += (size_t)B * 4 * H;
    float* hs    = ws + off; off += (size_t)B * H;
    float* cs    = ws + off; off += (size_t)B * H;
    float* qstar = ws + off; off += (size_t)B * 2 * H;
    float* rpool = ws + off; off += (size_t)B * H;
    float* score = ws + off; off += (size_t)N;
    float* ssum  = ws + off; off += (size_t)B;
    unsigned* smax = (unsigned*)(ws + off); off += (size_t)B;
    float* g1    = ws + off; off += (size_t)B * H;
    (void)ws_size; (void)n_in; (void)out_size;

    // -------- packed bf16 weight pool (32B-aligned region after floats) ----
    off = (off + 63) & ~(size_t)63;
    bf16_t* bfpool = (bf16_t*)(ws + off);
    size_t boff = 0;
    auto psize = [](int Nd, int K) {
        return (size_t)(Nd / 16) * ((K + 31) / 32) * 32 * 16;
    };
    auto balloc = [&](int Nd, int K) {
        bf16_t* p = bfpool + boff; boff += psize(Nd, K); return p;
    };
    bf16_t* pk_Wnode = balloc(H, FEAT);
    bf16_t* pk_Wedge = balloc(H, EDGE);
    bf16_t* pk_W1[3]; for (int l = 0; l < L; ++l) pk_W1[l] = balloc(H, H);
    bf16_t* pk_W2[3]; for (int l = 0; l < L; ++l) pk_W2[l] = balloc(H, H);
    bf16_t* pk_Wq    = balloc(H, H);
    bf16_t* pk_Wk    = balloc(H, H);
    bf16_t* pk_Wv    = balloc(H, H);
    bf16_t* pk_Wskip = balloc(H, H);
    bf16_t* pk_We    = balloc(H, H);
    bf16_t* pk_Wih   = balloc(4 * H, 2 * H);   // transposed pack (A@W^T)
    bf16_t* pk_Whh   = balloc(4 * H, H);       // transposed pack
    bf16_t* pk_hW1   = balloc(H, 2 * H);

    auto pack = [&](const float* W, bf16_t* outp, int Nd, int K, int transb) {
        long total = (long)(Nd / 16) * ((K + 31) / 32) * 32;
        pack_b_kernel<<<cdiv(total, 256), 256, 0, stream>>>(W, outp, Nd, K, transb, total);
    };
    pack(W_node, pk_Wnode, H, FEAT, 0);
    pack(W_edge, pk_Wedge, H, EDGE, 0);
    for (int l = 0; l < L; ++l) {
        pack(mlp_W1 + (size_t)l * H * H, pk_W1[l], H, H, 0);
        pack(mlp_W2 + (size_t)l * H * H, pk_W2[l], H, H, 0);
    }
    pack(Wq, pk_Wq, H, H, 0);
    pack(Wk, pk_Wk, H, H, 0);
    pack(Wv, pk_Wv, H, H, 0);
    pack(Wskip, pk_Wskip, H, H, 0);
    pack(We, pk_We, H, H, 0);
    pack(lstm_Wih, pk_Wih, 4 * H, 2 * H, 1);
    pack(lstm_Whh, pk_Whh, 4 * H, H, 1);
    pack(head_W1, pk_hW1, H, 2 * H, 0);

    const int gNodeB = N / 16;   // one block per 16-row panel (Nd=128 kernels)
    const int gEdgeB = E / 16;
    const int gHeadB = B / 16;
    const int gB512  = cdiv((long)(B / 16) * (512 / 16), 8);

    // -------- input projections --------
    wmma_gemm128_kernel<0, 64, false><<<gNodeB, 256, 0, stream>>>(
        x, nullptr, pk_Wnode, b_node, h);
    wmma_gemm128_kernel<0, 16, false><<<gEdgeB, 256, 0, stream>>>(
        eattr, nullptr, pk_Wedge, b_edge, ebuf);

    // -------- GINE layers --------
    const long totEH = (long)E * H;
    for (int l = 0; l < L; ++l) {
        hipMemsetAsync(agg, 0, NH * sizeof(float), stream);
        gine_msg_kernel<<<cdiv(totEH, 256), 256, 0, stream>>>(h, ebuf, src, dst, agg, totEH);
        wmma_gemm128_kernel<1, 128, true><<<gNodeB, 256, 0, stream>>>(
            h, agg, pk_W1[l], mlp_b1 + (size_t)l * H, t1);
        wmma_gemm128_kernel<0, 128, false><<<gNodeB, 256, 0, stream>>>(
            t1, nullptr, pk_W2[l], mlp_b2 + (size_t)l * H, zz);
        ln_gelu_res_kernel<<<cdiv(N, 8), 256, 0, stream>>>(
            zz, ln_g + (size_t)l * H, ln_b + (size_t)l * H, h, N);
    }

    // -------- TransformerConv --------
    wmma_gemm128_kernel<0, 128, false><<<gNodeB, 256, 0, stream>>>(h, nullptr, pk_Wq, bq, qb);
    wmma_gemm128_kernel<0, 128, false><<<gNodeB, 256, 0, stream>>>(h, nullptr, pk_Wk, bk, kb);
    wmma_gemm128_kernel<0, 128, false><<<gNodeB, 256, 0, stream>>>(h, nullptr, pk_Wv, bv, vb);
    wmma_gemm128_kernel<0, 128, false><<<gNodeB, 256, 0, stream>>>(h, nullptr, pk_Wskip, bskip, rb);
    wmma_gemm128_kernel<0, 128, false><<<gEdgeB, 256, 0, stream>>>(ebuf, nullptr, pk_We, be, eeb);

    fill_u32_kernel<<<cdiv((long)N * HEADS, 256), 256, 0, stream>>>(amax, NEG_INF_ORD, (long)N * HEADS);
    hipMemsetAsync(denom, 0, (size_t)N * HEADS * sizeof(float), stream);
    hipMemsetAsync(agg, 0, NH * sizeof(float), stream);  // attention output accumulator

    attn_alpha_kernel<<<cdiv(E, 8), 256, 0, stream>>>(qb, kb, eeb, src, dst, alpha, amax, E);
    attn_exp_kernel<<<cdiv((long)E * HEADS, 256), 256, 0, stream>>>(alpha, amax, denom, dst, (long)E * HEADS);
    attn_agg_kernel<<<cdiv(totEH, 256), 256, 0, stream>>>(alpha, denom, vb, eeb, src, dst, agg, totEH);
    beta_ln_kernel<<<cdiv(N, 8), 256, 0, stream>>>(agg, rb, Wbeta, tln_g, tln_b, h, N);

    // -------- Set2Set readout --------
    hipMemsetAsync(hs, 0, (size_t)B * H * sizeof(float), stream);
    hipMemsetAsync(cs, 0, (size_t)B * H * sizeof(float), stream);
    hipMemsetAsync(qstar, 0, (size_t)B * 2 * H * sizeof(float), stream);

    for (int s = 0; s < STEPS; ++s) {
        wmma_gemm_kernel<0, false, 256><<<gB512, 256, 0, stream>>>(
            qstar, pk_Wih, lstm_bih, gates, B, 4 * H);
        wmma_gemm_kernel<0, true, 128><<<gB512, 256, 0, stream>>>(
            hs, pk_Whh, lstm_bhh, gates, B, 4 * H);
        lstm_update_kernel<<<cdiv((long)B * H, 256), 256, 0, stream>>>(gates, cs, hs, B * H);

        fill_u32_kernel<<<cdiv(B, 256), 256, 0, stream>>>(smax, NEG_INF_ORD, B);
        hipMemsetAsync(ssum, 0, B * sizeof(float), stream);
        hipMemsetAsync(rpool, 0, (size_t)B * H * sizeof(float), stream);

        s2s_score_kernel<<<cdiv(N, 8), 256, 0, stream>>>(h, hs, batch, score, smax, N);
        s2s_exp_kernel<<<cdiv(N, 256), 256, 0, stream>>>(score, smax, ssum, batch, N);
        s2s_pool_kernel<<<cdiv((long)N * H, 256), 256, 0, stream>>>(score, ssum, h, batch, rpool, (long)N * H);
        pack_qstar_kernel<<<cdiv((long)B * 2 * H, 256), 256, 0, stream>>>(hs, rpool, qstar, B * 2 * H);
    }

    // -------- MLP head --------
    wmma_gemm128_kernel<1, 256, false><<<gHeadB, 256, 0, stream>>>(
        qstar, nullptr, pk_hW1, head_b1, g1);
    head_out_kernel<<<cdiv(B, 256), 256, 0, stream>>>(g1, head_W2, head_b2, (float*)d_out, B);
}
